// LSTMCRF_85057532330085
// MI455X (gfx1250) — matrix operand order repaired and checked by reading, not compiled, not run on hardware
//
#include <hip/hip_runtime.h>

typedef __attribute__((ext_vector_type(16))) _Float16 v16h;
typedef __attribute__((ext_vector_type(8)))  float    v8f;
typedef __attribute__((ext_vector_type(2)))  _Float16 h2;

#define TLEN  512
#define KTAG  32
#define NWAVE 8
#define CHUNK (TLEN / NWAVE)   // 64 steps per wave
#define BOSI  30
#define EOSI  31

__device__ __forceinline__ float h2_as_f(h2 x)   { return __builtin_bit_cast(float, x); }
__device__ __forceinline__ h2    f_as_h2(float x){ return __builtin_bit_cast(h2, x); }
// __builtin_amdgcn_cvt_pkrtz returns __fp16x2; bit-cast to our types
__device__ __forceinline__ h2    pkrtz(float a, float b) {
    return __builtin_bit_cast(h2, __builtin_amdgcn_cvt_pkrtz(a, b));
}
__device__ __forceinline__ float pkrtz_f(float a, float b) {
    return __builtin_bit_cast(float, __builtin_amdgcn_cvt_pkrtz(a, b));
}

// xor-16 half-wave swap on the VALU (no LDS, no DScnt wait)
__device__ __forceinline__ float permx16(float x) {
    int i = __builtin_bit_cast(int, x);
    int r = __builtin_amdgcn_permlanex16(i, i, 0x76543210, 0xfedcba98, false, false);
    return __builtin_bit_cast(float, r);
}
// DPP16 row rotate-right by N (N in 1..15), VALU lane permute within rows of 16
template <int N>
__device__ __forceinline__ float row_ror(float x) {
    int i = __builtin_bit_cast(int, x);
    int r = __builtin_amdgcn_update_dpp(i, i, 0x120 | N, 0xf, 0xf, false);
    return __builtin_bit_cast(float, r);
}
// wave-wide max using only VALU lane permutes
__device__ __forceinline__ float wave_max(float m) {
    m = fmaxf(m, permx16(m));      // fold upper 16 lanes into lower (and vice versa)
    m = fmaxf(m, row_ror<8>(m));   // rotation-based covering reduction within 16
    m = fmaxf(m, row_ror<4>(m));
    m = fmaxf(m, row_ror<2>(m));
    m = fmaxf(m, row_ror<1>(m));
    return m;
}

__global__ __launch_bounds__(256)
void crf_fwd_wmma_kernel(const float* __restrict__ emission,
                         const float* __restrict__ trans,
                         float* __restrict__ out)
{
    // per-wave chunk products (col-major: row + 32*col) + per-chunk log2 scale
    __shared__ float sG[NWAVE][KTAG * KTAG];
    __shared__ float sLog[NWAVE];

    const int  b    = blockIdx.x;
    const int  tid  = threadIdx.x;
    const int  w    = tid >> 5;      // wave id (wave32)
    const int  lane = tid & 31;
    const int  lr   = lane & 15;
    const bool lo   = (lane < 16);
    const int  kb   = lo ? 0 : 8;    // K-base of this lane half in 16-bit A layout

    // ---------------- one-time: E = exp(trans) in WMMA A-layout (f16, 2 M-tiles) ----
    // A 16x32 layout used here (contiguous k): lanes 0-15 hold K {0..15}, lanes 16-31 hold {16..31};
    // VGPR v holds the pair (k0, k0+1), k0 = (lo ? 0 : 16) + 2v. Row = 16*tile + (lane&15).
    v16h EA0, EA1;
    #pragma unroll
    for (int v = 0; v < 8; ++v) {
        const int k0 = (lo ? 0 : 16) + 2 * v;
        h2 p0 = pkrtz(__expf(trans[lr * KTAG + k0]),
                      __expf(trans[lr * KTAG + k0 + 1]));
        h2 p1 = pkrtz(__expf(trans[(16 + lr) * KTAG + k0]),
                      __expf(trans[(16 + lr) * KTAG + k0 + 1]));
        EA0[2 * v] = p0[0]; EA0[2 * v + 1] = p0[1];
        EA1[2 * v] = p1[0]; EA1[2 * v + 1] = p1[1];
    }

    // ---------------- accumulator in B-layout (f16, 2 N-tiles) = identity ----------
    // B 32x16, contiguous k: lanes 0-15: K = 2v,2v+1; lanes 16-31: K = 16+2v,16+2v+1; col = 16n + (lane&15)
    v16h B0, B1;
    #pragma unroll
    for (int v = 0; v < 8; ++v) {
        const int k0 = lo ? (2 * v) : (16 + 2 * v);
        B0[2 * v]     = (_Float16)((k0     == lr     ) ? 1.f : 0.f);
        B0[2 * v + 1] = (_Float16)((k0 + 1 == lr     ) ? 1.f : 0.f);
        B1[2 * v]     = (_Float16)((k0     == 16 + lr) ? 1.f : 0.f);
        B1[2 * v + 1] = (_Float16)((k0 + 1 == 16 + lr) ? 1.f : 0.f);
    }

    const float* ep = emission + ((size_t)b * TLEN + (size_t)w * CHUNK) * KTAG + lane;
    float ev    = ep[0];     // e_t[lane], prefetched
    float inv_s = 1.0f;      // 2^-k rescale folded into next step's diag(w)
    float logk  = 0.0f;      // accumulated exponent (units of ln2)
    v8f d00 = {}, d01 = {}, d10 = {}, d11 = {};

    for (int t = 0; t < CHUNK; ++t) {
        const int tn  = (t + 1 < CHUNK) ? (t + 1) : (CHUNK - 1);
        const float evn = ep[tn * KTAG];                       // prefetch next e_t

        // row weights w[j] = exp(e_t[j]) * inv_s, broadcast per A-row via permlanex16
        const float we  = __expf(ev) * inv_s;
        const float wo  = permx16(we);                         // other half's value
        const float wa0 = lo ? we : wo;                        // rows 0..15
        const float wa1 = lo ? wo : we;                        // rows 16..31
        h2 hh0; hh0[0] = (_Float16)wa0; hh0[1] = (_Float16)wa0;
        h2 hh1; hh1[0] = (_Float16)wa1; hh1[1] = (_Float16)wa1;
        const float wf0 = h2_as_f(hh0), wf1 = h2_as_f(hh1);
        v8f t0, t1;
        #pragma unroll
        for (int r = 0; r < 8; ++r) { t0[r] = wf0; t1[r] = wf1; }
        const v16h A0 = EA0 * __builtin_bit_cast(v16h, t0);    // v_pk_mul_f16 x8
        const v16h A1 = EA1 * __builtin_bit_cast(v16h, t1);

        // D(32x32,f32) = A(diag(w)E) x B(accumulator): 4 WMMAs, K=32 in one shot
        const v8f zc = {};
        d00 = __builtin_amdgcn_wmma_f32_16x16x32_f16(false, A0, false, B0, (short)0, zc, false, false);
        d01 = __builtin_amdgcn_wmma_f32_16x16x32_f16(false, A0, false, B1, (short)0, zc, false, false);
        d10 = __builtin_amdgcn_wmma_f32_16x16x32_f16(false, A1, false, B0, (short)0, zc, false, false);
        d11 = __builtin_amdgcn_wmma_f32_16x16x32_f16(false, A1, false, B1, (short)0, zc, false, false);

        // exact power-of-two rescale from matrix max (one scalar per step)
        float mx = 0.f;
        #pragma unroll
        for (int r = 0; r < 8; ++r)
            mx = fmaxf(fmaxf(fmaxf(mx, d00[r]), fmaxf(d01[r], d10[r])), d11[r]);
        mx = wave_max(mx);
        const unsigned eb = (__float_as_uint(mx) >> 23) & 0xffu;
        inv_s = __uint_as_float((254u - eb) << 23);            // 2^-(eb-127), exact
        logk += (float)((int)eb - 127);

        // D-layout(f32) -> B-layout(f16): pack row pairs, then one xor-16 swap + select
        float p00[4], p01[4], p10[4], p11[4];
        #pragma unroll
        for (int v = 0; v < 4; ++v) {
            p00[v] = pkrtz_f(d00[2 * v], d00[2 * v + 1]);
            p01[v] = pkrtz_f(d01[2 * v], d01[2 * v + 1]);
            p10[v] = pkrtz_f(d10[2 * v], d10[2 * v + 1]);
            p11[v] = pkrtz_f(d11[2 * v], d11[2 * v + 1]);
        }
        #pragma unroll
        for (int v = 0; v < 4; ++v) {
            const float q00 = permx16(p00[v]);
            const float q10 = permx16(p10[v]);
            const float q01 = permx16(p01[v]);
            const float q11 = permx16(p11[v]);
            const h2 b_lo = f_as_h2(lo ? p00[v] : q10);        // B0 regs v   (K 0..7 / 16..23)
            const h2 b_hi = f_as_h2(lo ? q00    : p10[v]);     // B0 regs v+4 (K 8..15 / 24..31)
            const h2 c_lo = f_as_h2(lo ? p01[v] : q11);
            const h2 c_hi = f_as_h2(lo ? q01    : p11[v]);
            B0[2 * v]           = b_lo[0]; B0[2 * v + 1]           = b_lo[1];
            B0[2 * (v + 4)]     = b_hi[0]; B0[2 * (v + 4) + 1]     = b_hi[1];
            B1[2 * v]           = c_lo[0]; B1[2 * v + 1]           = c_lo[1];
            B1[2 * (v + 4)]     = c_hi[0]; B1[2 * (v + 4) + 1]     = c_hi[1];
        }
        ev = evn;
    }

    // store normalized chunk matrix to LDS (col-major), plus its log2 scale count
    #pragma unroll
    for (int r = 0; r < 8; ++r) {
        const int rr = r + (lo ? 0 : 8);
        sG[w][rr        + KTAG * lr       ] = d00[r] * inv_s;
        sG[w][rr        + KTAG * (16 + lr)] = d01[r] * inv_s;
        sG[w][(16 + rr) + KTAG * lr       ] = d10[r] * inv_s;
        sG[w][(16 + rr) + KTAG * (16 + lr)] = d11[r] * inv_s;
    }
    if (lane == 0) sLog[w] = logk;
    __syncthreads();

    // ---------------- combine: v = G7 ... G0 * e_BOS, then dot with exp(trans[EOS,:]) ----
    if (w == 0) {
        float v    = (lane == BOSI) ? 1.f : 0.f;
        float ktot = 0.f;
        #pragma unroll 1
        for (int c = 0; c < NWAVE; ++c) {
            float acc = 0.f;
            #pragma unroll
            for (int i = 0; i < KTAG; ++i)
                acc = fmaf(sG[c][lane + KTAG * i], __shfl(v, i, 32), acc);
            float mx = wave_max(acc);
            const unsigned eb = (__float_as_uint(mx) >> 23) & 0xffu;
            v = acc * __uint_as_float((254u - eb) << 23);
            ktot += (float)((int)eb - 127) + sLog[c];
        }
        float term = __expf(trans[EOSI * KTAG + lane]) * v;
        #pragma unroll
        for (int off = 16; off >= 1; off >>= 1)
            term += __shfl_xor(term, off, 32);
        if (lane == 0)
            out[b] = __logf(term) + 0.69314718055994531f * ktot;
    }
}

extern "C" void kernel_launch(void* const* d_in, const int* in_sizes, int n_in,
                              void* d_out, int out_size, void* d_ws, size_t ws_size,
                              hipStream_t stream) {
    const float* emission = (const float*)d_in[0];
    const float* trans    = (const float*)d_in[1];
    float*       out      = (float*)d_out;
    const int nb = in_sizes[0] / (TLEN * KTAG);   // batch = 2048
    crf_fwd_wmma_kernel<<<nb, NWAVE * 32, 0, stream>>>(emission, trans, out);
}
